// ReGRU_18940805775609
// MI455X (gfx1250) — matrix operand order repaired and checked by reading, not compiled, not run on hardware
//
#include <hip/hip_runtime.h>
#include <hip/hip_bf16.h>

#define H  512
#define BB 1024
#define TT 128
#define H3 (3 * H)

typedef __attribute__((ext_vector_type(16))) __bf16       v16bf;
typedef __attribute__((ext_vector_type(8)))  float        v8f;
typedef __attribute__((ext_vector_type(4)))  unsigned int v4u;

union ABFrag { v16bf v; v4u q[2]; };

// A-fragment (16x32 bf16). Per-lane row pointer pre-offset by half*8:
// elems 0..7 -> K = k+half*8+0..7 ; elems 8..15 -> K = k+16+half*8+0..7
__device__ __forceinline__ v16bf ld_a(const __bf16* p, int k) {
    ABFrag u;
    u.q[0] = *(const v4u*)(p + k);
    u.q[1] = *(const v4u*)(p + k + 16);
    return u.v;
}

// B-fragment (32x16 bf16). Per-lane row pointer pre-offset by half*8:
// elems 0..7 -> K = k+half*8+0..7 ; elems 8..15 -> K = k+16+half*8+0..7 at N = lane&15.
__device__ __forceinline__ v16bf ld_b(const __bf16* p, int k) {
    ABFrag u;
    u.q[0] = *(const v4u*)(p + k);
    u.q[1] = *(const v4u*)(p + k + 16);
    return u.v;
}

__device__ __forceinline__ v8f wmma_bf16(v16bf a, v16bf b, v8f c) {
    return __builtin_amdgcn_wmma_f32_16x16x32_bf16(false, a, false, b, (short)0, c,
                                                   false, false);
}

// Fused GRU cell. Per-wave 32x32 output tile of h'.
// LAYER==0: input term is a rank-1 outer product (input size 1), computed
//           analytically in the epilogue; only h@w_hh^T runs on WMMA.
// LAYER==1: r/z gates accumulate BOTH GEMMs into one accumulator
//           (only their sum is needed); n gate keeps i/h parts separate.
template <int LAYER>
__global__ __launch_bounds__(256) void gru_cell_kernel(
    const __bf16* __restrict__ hin_bf,   // [B,H] previous hidden, bf16
    const float*  __restrict__ hin_f,    // [B,H] previous hidden, f32
    const __bf16* __restrict__ xin_bf,   // LAYER==1: [B,H] input (h0 new), bf16
    const float*  __restrict__ xin_sc,   // LAYER==0: scalar input stream
    int xstride, int xoff,
    const __bf16* __restrict__ w_hh,     // [3H,H] bf16
    const __bf16* __restrict__ w_ih,     // LAYER==1: [3H,H] bf16
    const float*  __restrict__ w_ih0,    // LAYER==0: [3H,1] f32
    const float*  __restrict__ b_ih,
    const float*  __restrict__ b_hh,
    float*  __restrict__ hout_f,
    __bf16* __restrict__ hout_bf)
{
    const int lane = threadIdx.x & 31;
    const int wave = threadIdx.x >> 5;
    const int lo   = lane & 15;
    const int half = lane >> 4;
    const int m0 = blockIdx.y * 128 + (wave >> 1) * 32;  // 4 row-waves x 32 rows
    const int n0 = blockIdx.x * 64  + (wave & 1) * 32;   // 2 col-waves x 32 cols

    const v8f vz = {0.f, 0.f, 0.f, 0.f, 0.f, 0.f, 0.f, 0.f};
    v8f acc0[3][2][2];                 // LAYER==0: [gate][ntile][mtile]
    v8f accRZ[2][2][2];                // LAYER==1: [r/z][ntile][mtile] (i+h sum)
    v8f accNi[2][2], accNh[2][2];      // LAYER==1: n-gate i / h parts
#pragma unroll
    for (int t = 0; t < 2; ++t)
#pragma unroll
        for (int mt = 0; mt < 2; ++mt) {
            acc0[0][t][mt] = vz; acc0[1][t][mt] = vz; acc0[2][t][mt] = vz;
            accRZ[0][t][mt] = vz; accRZ[1][t][mt] = vz;
            accNi[t][mt] = vz; accNh[t][mt] = vz;
        }

    // Hoisted per-lane base pointers (constant-offset loads inside the loop).
    const __bf16* pAH[2];
    const __bf16* pAX[2];
#pragma unroll
    for (int mt = 0; mt < 2; ++mt) {
        pAH[mt] = hin_bf + (size_t)(m0 + mt * 16 + lo) * H + half * 8;
        pAX[mt] = (LAYER == 1) ? (xin_bf + (size_t)(m0 + mt * 16 + lo) * H + half * 8)
                               : nullptr;
    }
    const __bf16* pBH[3][2];
    const __bf16* pBI[3][2];
#pragma unroll
    for (int g = 0; g < 3; ++g)
#pragma unroll
        for (int t = 0; t < 2; ++t) {
            const int ncol = g * H + n0 + t * 16 + lo;
            pBH[g][t] = w_hh + (size_t)ncol * H + half * 8;
            pBI[g][t] = (LAYER == 1) ? (w_ih + (size_t)ncol * H + half * 8) : nullptr;
        }

#pragma unroll
    for (int kk = 0; kk < H / 32; ++kk) {
        const int k = kk * 32;
        // ---- all loads for this K-slice first ----
        v16bf fAH[2], fAX[2], fBH[3][2], fBI[3][2];
#pragma unroll
        for (int mt = 0; mt < 2; ++mt) {
            fAH[mt] = ld_a(pAH[mt], k);
            if (LAYER == 1) fAX[mt] = ld_a(pAX[mt], k);
        }
#pragma unroll
        for (int g = 0; g < 3; ++g)
#pragma unroll
            for (int t = 0; t < 2; ++t) {
                fBH[g][t] = ld_b(pBH[g][t], k);
                if (LAYER == 1) fBI[g][t] = ld_b(pBI[g][t], k);
            }
        // ---- matrix ops ----
#pragma unroll
        for (int t = 0; t < 2; ++t)
#pragma unroll
            for (int mt = 0; mt < 2; ++mt) {
                if (LAYER == 0) {
                    acc0[0][t][mt] = wmma_bf16(fAH[mt], fBH[0][t], acc0[0][t][mt]);
                    acc0[1][t][mt] = wmma_bf16(fAH[mt], fBH[1][t], acc0[1][t][mt]);
                    acc0[2][t][mt] = wmma_bf16(fAH[mt], fBH[2][t], acc0[2][t][mt]);
                } else {
                    accRZ[0][t][mt] = wmma_bf16(fAH[mt], fBH[0][t],
                                      wmma_bf16(fAX[mt], fBI[0][t], accRZ[0][t][mt]));
                    accRZ[1][t][mt] = wmma_bf16(fAH[mt], fBH[1][t],
                                      wmma_bf16(fAX[mt], fBI[1][t], accRZ[1][t][mt]));
                    accNi[t][mt] = wmma_bf16(fAX[mt], fBI[2][t], accNi[t][mt]);
                    accNh[t][mt] = wmma_bf16(fAH[mt], fBH[2][t], accNh[t][mt]);
                }
            }
    }

    // Epilogue. Acc element v of tile (t,mt): M = m0+mt*16+half*8+v, N = n0+t*16+lo
#pragma unroll
    for (int mt = 0; mt < 2; ++mt)
#pragma unroll
        for (int t = 0; t < 2; ++t)
#pragma unroll
            for (int v = 0; v < 8; ++v) {
                const int m   = m0 + mt * 16 + half * 8 + v;
                const int col = n0 + t * 16 + lo;
                float sr, sz, in_, hn;
                if (LAYER == 0) {
                    const float xv = xin_sc[(size_t)m * xstride + xoff];
                    const float ir = xv * w_ih0[col]         + b_ih[col];
                    const float iz = xv * w_ih0[H + col]     + b_ih[H + col];
                    in_ = xv * w_ih0[2 * H + col] + b_ih[2 * H + col];
                    sr = ir + acc0[0][t][mt][v] + b_hh[col];
                    sz = iz + acc0[1][t][mt][v] + b_hh[H + col];
                    hn = acc0[2][t][mt][v] + b_hh[2 * H + col];
                } else {
                    sr  = accRZ[0][t][mt][v] + b_ih[col]     + b_hh[col];
                    sz  = accRZ[1][t][mt][v] + b_ih[H + col] + b_hh[H + col];
                    in_ = accNi[t][mt][v] + b_ih[2 * H + col];
                    hn  = accNh[t][mt][v] + b_hh[2 * H + col];
                }
                const float r  = 1.f / (1.f + __expf(-sr));
                const float z  = 1.f / (1.f + __expf(-sz));
                const float n  = tanhf(in_ + r * hn);
                const float ho = hin_f[(size_t)m * H + col];
                const float hv = (1.f - z) * n + z * ho;
                hout_f[(size_t)m * H + col]  = hv;
                hout_bf[(size_t)m * H + col] = (__bf16)hv;
            }
}

// pred[b] = dot(h1[b,:], w_lin) + b_lin  -> out[b*T + t] and feedback buffer
__global__ __launch_bounds__(256) void head_kernel(
    const float* __restrict__ h1, const float* __restrict__ w_lin,
    const float* __restrict__ b_lin, float* __restrict__ out, int tstep,
    float* __restrict__ pred)
{
    const int lane = threadIdx.x & 31;
    const int wave = threadIdx.x >> 5;
    const int row  = blockIdx.x * 8 + wave;
    const float* hr = h1 + (size_t)row * H;
    float s = 0.f;
#pragma unroll
    for (int i = 0; i < H / 32; ++i) s += hr[lane + i * 32] * w_lin[lane + i * 32];
#pragma unroll
    for (int off = 16; off > 0; off >>= 1) s += __shfl_xor(s, off, 32);
    if (lane == 0) {
        const float v = s + b_lin[0];
        out[(size_t)row * TT + tstep] = v;
        pred[row] = v;
    }
}

__global__ void cvt_bf16_kernel(const float* __restrict__ src,
                                __bf16* __restrict__ dst, int n) {
    int i = blockIdx.x * blockDim.x + threadIdx.x;
    if (i < n) dst[i] = (__bf16)src[i];
}

__global__ void zero_kernel(unsigned int* __restrict__ p, int n) {
    int i = blockIdx.x * blockDim.x + threadIdx.x;
    if (i < n) p[i] = 0u;
}

extern "C" void kernel_launch(void* const* d_in, const int* in_sizes, int n_in,
                              void* d_out, int out_size, void* d_ws, size_t ws_size,
                              hipStream_t stream) {
    (void)in_sizes; (void)n_in; (void)out_size; (void)ws_size;
    const float* x     = (const float*)d_in[0];   // [B,T,1]
    const float* w_ih0 = (const float*)d_in[3];   // [3H,1]
    const float* w_hh0 = (const float*)d_in[4];   // [3H,H]
    const float* b_ih0 = (const float*)d_in[5];
    const float* b_hh0 = (const float*)d_in[6];
    const float* w_ih1 = (const float*)d_in[7];   // [3H,H]
    const float* w_hh1 = (const float*)d_in[8];   // [3H,H]
    const float* b_ih1 = (const float*)d_in[9];
    const float* b_hh1 = (const float*)d_in[10];
    const float* w_lin = (const float*)d_in[11];  // [1,H]
    const float* b_lin = (const float*)d_in[12];
    float* out = (float*)d_out;                   // [B,T,1]

    char*  ws  = (char*)d_ws;
    size_t off = 0;
    auto alloc = [&](size_t bytes) {
        char* p = ws + off;
        off += (bytes + 255) & ~(size_t)255;
        return p;
    };
    __bf16* whh0b = (__bf16*)alloc((size_t)H3 * H * 2);
    __bf16* wih1b = (__bf16*)alloc((size_t)H3 * H * 2);
    __bf16* whh1b = (__bf16*)alloc((size_t)H3 * H * 2);
    float*  h0f[2] = {(float*)alloc((size_t)BB * H * 4), (float*)alloc((size_t)BB * H * 4)};
    float*  h1f[2] = {(float*)alloc((size_t)BB * H * 4), (float*)alloc((size_t)BB * H * 4)};
    __bf16* h0b[2] = {(__bf16*)alloc((size_t)BB * H * 2), (__bf16*)alloc((size_t)BB * H * 2)};
    __bf16* h1b[2] = {(__bf16*)alloc((size_t)BB * H * 2), (__bf16*)alloc((size_t)BB * H * 2)};
    float*  pred   = (float*)alloc((size_t)BB * 4);

    const int NW = H3 * H;
    cvt_bf16_kernel<<<(NW + 255) / 256, 256, 0, stream>>>(w_hh0, whh0b, NW);
    cvt_bf16_kernel<<<(NW + 255) / 256, 256, 0, stream>>>(w_ih1, wih1b, NW);
    cvt_bf16_kernel<<<(NW + 255) / 256, 256, 0, stream>>>(w_hh1, whh1b, NW);
    zero_kernel<<<(BB * H + 255) / 256, 256, 0, stream>>>((unsigned int*)h0f[0], BB * H);
    zero_kernel<<<(BB * H + 255) / 256, 256, 0, stream>>>((unsigned int*)h1f[0], BB * H);
    zero_kernel<<<(BB * H / 2 + 255) / 256, 256, 0, stream>>>((unsigned int*)h0b[0], BB * H / 2);
    zero_kernel<<<(BB * H / 2 + 255) / 256, 256, 0, stream>>>((unsigned int*)h1b[0], BB * H / 2);

    dim3 grid(H / 64, BB / 128);  // (8, 8): col-blocks x row-blocks
    int cur = 0;

    // Phase 1: teacher-forced scan over x
    for (int t = 0; t < TT; ++t) {
        const int nxt = cur ^ 1;
        gru_cell_kernel<0><<<grid, 256, 0, stream>>>(
            h0b[cur], h0f[cur], nullptr, x, TT, t,
            whh0b, nullptr, w_ih0, b_ih0, b_hh0, h0f[nxt], h0b[nxt]);
        gru_cell_kernel<1><<<grid, 256, 0, stream>>>(
            h1b[cur], h1f[cur], h0b[nxt], nullptr, 0, 0,
            whh1b, wih1b, nullptr, b_ih1, b_hh1, h1f[nxt], h1b[nxt]);
        cur = nxt;
    }
    head_kernel<<<BB / 8, 256, 0, stream>>>(h1f[cur], w_lin, b_lin, out, 0, pred);

    // Phase 2: autoregressive, feed scalar prediction back
    for (int t = 1; t < TT; ++t) {
        const int nxt = cur ^ 1;
        gru_cell_kernel<0><<<grid, 256, 0, stream>>>(
            h0b[cur], h0f[cur], nullptr, pred, 1, 0,
            whh0b, nullptr, w_ih0, b_ih0, b_hh0, h0f[nxt], h0b[nxt]);
        gru_cell_kernel<1><<<grid, 256, 0, stream>>>(
            h1b[cur], h1f[cur], h0b[nxt], nullptr, 0, 0,
            whh1b, wih1b, nullptr, b_ih1, b_hh1, h1f[nxt], h1b[nxt]);
        cur = nxt;
        head_kernel<<<BB / 8, 256, 0, stream>>>(h1f[cur], w_lin, b_lin, out, t, pred);
    }
}
